// FourierMappedINR_83717502534140
// MI455X (gfx1250) — compile-verified, matrix operand order checked by reading, not run on hardware
//
#include <hip/hip_runtime.h>

typedef __attribute__((ext_vector_type(16))) _Float16 v16h;
typedef __attribute__((ext_vector_type(8)))  float    v8f;
typedef unsigned int u32;

#define HRW 1280
#define HRH 720
#define LRW 640
#define LRH 360
#define NPIX_HR (HRW*HRH)
#define NPIX_LR (LRW*LRH)

union Frag16 { v16h h; u32 u[8]; };

// 16-bit A-matrix 16x32 VGPR layout (ISA 7.12.2): VGPR g holds K pair starting at
// k = 2*(g&3) + 16*(g>>2) + 8*(lane>=16). B (32x16) mirrors with lane = N column.
__device__ __forceinline__ int kpair(int g, int hi) {
  return 2*(g & 3) + ((g >> 2) << 4) + (hi << 3);
}

__device__ __forceinline__ v16h load_frag(const _Float16* row, int hi) {
  Frag16 f;
#pragma unroll
  for (int g = 0; g < 8; ++g)
    f.u[g] = *(const u32*)(row + kpair(g, hi));
  return f.h;
}

__device__ __forceinline__ v8f wmma_f16(v16h a, v16h b, v8f c) {
  return __builtin_amdgcn_wmma_f32_16x16x32_f16(false, a, false, b, (short)0, c, false, false);
}

__device__ __forceinline__ u32 pack2h(float a, float b) {
  union { _Float16 h[2]; u32 u; } p;
  p.h[0] = (_Float16)a; p.h[1] = (_Float16)b;
  return p.u;
}

// ---------------- NCHW f32 -> (H,W,C=32) f16 transpose ----------------
__global__ __launch_bounds__(256) void nchw_to_hwc_f16(const float* __restrict__ src,
                                                       _Float16* __restrict__ dst,
                                                       int npix) {
  int px = blockIdx.x * 256 + threadIdx.x;
  if (px >= npix) return;
  union { _Float16 h[32]; uint4 q[4]; } buf;
#pragma unroll
  for (int c = 0; c < 32; ++c)
    buf.h[c] = (_Float16)src[(size_t)c * npix + px];
  uint4* d = (uint4*)(dst + (size_t)px * 32);
#pragma unroll
  for (int i = 0; i < 4; ++i) d[i] = buf.q[i];
}

// ---------------- weight repack to f16 WMMA-A-friendly tiles ----------------
__global__ __launch_bounds__(256) void prep_weights(const float* __restrict__ amp_w,
                                                    const float* __restrict__ freq_w,
                                                    const float* __restrict__ w0,
                                                    const float* __restrict__ w1,
                                                    const float* __restrict__ w2,
                                                    const float* __restrict__ wo,
                                                    _Float16* __restrict__ wA,
                                                    _Float16* __restrict__ wF,
                                                    _Float16* __restrict__ wL,
                                                    _Float16* __restrict__ wO) {
  int tid = threadIdx.x;
  // (64,32,3,3) -> [tap p][out o][in c]
  for (int i = tid; i < 9 * 64 * 32; i += 256) {
    int p = i / 2048, r = i % 2048, o = r / 32, c = r % 32;
    wF[i] = (_Float16)freq_w[o * 288 + c * 9 + p];
    wA[i] = (_Float16)amp_w[o * 288 + c * 9 + p];
  }
  for (int i = tid; i < 3 * 4096; i += 256) {
    int l = i / 4096, r = i % 4096;
    const float* w = (l == 0) ? w0 : (l == 1) ? w1 : w2;
    wL[i] = (_Float16)w[r];
  }
  for (int i = tid; i < 16 * 64; i += 256) {
    int m = i / 64, c = i % 64;
    wO[i] = (_Float16)((m < 12) ? wo[m * 64 + c] : 0.0f);
  }
}

// ---------------- LR 3x3 conv (32->64) via WMMA, output (H,W,64) f16 ----------------
__global__ __launch_bounds__(256) void amp_conv(const float* __restrict__ amp_b,
                                                const _Float16* __restrict__ lrT,
                                                const _Float16* __restrict__ wA,
                                                _Float16* __restrict__ ampT) {
  __shared__ __align__(16) unsigned char s_lr[180 * 80]; // 10 rows x 18 cols, 80B stride
  int tid = threadIdx.x;
  int bx = blockIdx.x % 40, by = blockIdx.x / 40;
  int x0 = bx * 16, y0 = by * 8;

  for (int i = tid; i < 180 * 4; i += 256) {
    int rec = i >> 2, part = i & 3;
    int pr = rec / 18, pc = rec - pr * 18;
    int gy = y0 - 1 + pr, gx = x0 - 1 + pc;
    uint4 v = {0u, 0u, 0u, 0u};
    if (gy >= 0 && gy < LRH && gx >= 0 && gx < LRW)
      v = *(const uint4*)(lrT + ((size_t)gy * LRW + gx) * 32 + part * 8);
    *(uint4*)(s_lr + rec * 80 + part * 16) = v;
  }
  __syncthreads();

  int wv = tid >> 5, ln = tid & 31, n = ln & 15, hi = ln >> 4;
  int y = y0 + wv, x = x0 + n;

  v8f acc[4];
#pragma unroll
  for (int mb = 0; mb < 4; ++mb)
#pragma unroll
    for (int j = 0; j < 8; ++j) acc[mb][j] = amp_b[mb * 16 + hi * 8 + j];

#pragma unroll
  for (int p = 0; p < 9; ++p) {
    int dy = p / 3, dx = p - dy * 3;
    const _Float16* brow = (const _Float16*)(s_lr + ((wv + dy) * 18 + (n + dx)) * 80);
    v16h bf = load_frag(brow, hi);
#pragma unroll
    for (int mb = 0; mb < 4; ++mb) {
      v16h af = load_frag(wA + ((size_t)p * 64 + mb * 16 + n) * 32, hi);
      acc[mb] = wmma_f16(af, bf, acc[mb]);
    }
  }

  size_t base = ((size_t)y * LRW + x) * 64;
#pragma unroll
  for (int mb = 0; mb < 4; ++mb)
#pragma unroll
    for (int jp = 0; jp < 4; ++jp)
      *(u32*)(ampT + base + mb * 16 + hi * 8 + 2 * jp) =
          pack2h(acc[mb][2 * jp], acc[mb][2 * jp + 1]);
}

// ------ fused HR pipeline: freq conv -> fourier -> 3x SIREN layers -> 12ch out ------
__global__ __launch_bounds__(256) void inr_pipeline(const float* __restrict__ freq_b,
                                                    const float* __restrict__ phase_w,
                                                    const float* __restrict__ b0,
                                                    const float* __restrict__ b1,
                                                    const float* __restrict__ b2,
                                                    const float* __restrict__ bo,
                                                    const _Float16* __restrict__ gbT,
                                                    const _Float16* __restrict__ ampT,
                                                    const _Float16* __restrict__ wF,
                                                    const _Float16* __restrict__ wL,
                                                    const _Float16* __restrict__ wO,
                                                    float* __restrict__ out) {
  __shared__ __align__(16) unsigned char s_gb[180 * 80];   // 10x18 halo, 80B/px
  __shared__ __align__(16) unsigned char s_amp[32 * 144];  // 4x8 LR px, 144B/px
  __shared__ __align__(16) unsigned char s_x[8 * 2176];    // per-wave 16px x 64ch, 136B/px

  int tid = threadIdx.x;
  int bx = blockIdx.x % 80, by = blockIdx.x / 80;
  int x0 = bx * 16, y0 = by * 8;

  for (int i = tid; i < 180 * 4; i += 256) {
    int rec = i >> 2, part = i & 3;
    int pr = rec / 18, pc = rec - pr * 18;
    int gy = y0 - 1 + pr, gx = x0 - 1 + pc;
    uint4 v = {0u, 0u, 0u, 0u};
    if (gy >= 0 && gy < HRH && gx >= 0 && gx < HRW)
      v = *(const uint4*)(gbT + ((size_t)gy * HRW + gx) * 32 + part * 8);
    *(uint4*)(s_gb + rec * 80 + part * 16) = v;
  }
  {
    int ly0 = y0 >> 1, lx0 = x0 >> 1;
    int i = tid; // exactly 256 chunks
    int rec = i >> 3, part = i & 7;
    int r = rec >> 3, c = rec & 7;
    uint4 v = *(const uint4*)(ampT + ((size_t)(ly0 + r) * LRW + (lx0 + c)) * 64 + part * 8);
    *(uint4*)(s_amp + rec * 144 + part * 16) = v;
  }
  __syncthreads();

  int wv = tid >> 5, ln = tid & 31, n = ln & 15, hi = ln >> 4;
  int y = y0 + wv, x = x0 + n;
  _Float16* xt = (_Float16*)(s_x + wv * 2176);

  // ---- freq 3x3 conv: D[64,16] via 9 taps x 4 m-blocks of wmma ----
  v8f acc[4];
#pragma unroll
  for (int mb = 0; mb < 4; ++mb)
#pragma unroll
    for (int j = 0; j < 8; ++j) acc[mb][j] = freq_b[mb * 16 + hi * 8 + j];

#pragma unroll
  for (int p = 0; p < 9; ++p) {
    int dy = p / 3, dx = p - dy * 3;
    const _Float16* brow = (const _Float16*)(s_gb + ((wv + dy) * 18 + (n + dx)) * 80);
    v16h bf = load_frag(brow, hi);
#pragma unroll
    for (int mb = 0; mb < 4; ++mb) {
      v16h af = load_frag(wF + ((size_t)p * 64 + mb * 16 + n) * 32, hi);
      acc[mb] = wmma_f16(af, bf, acc[mb]);
    }
  }

  // ---- fourier mapping + amplitude modulation -> x0 tile in LDS ----
  const float PI_F = 3.14159265358979323846f;
  float sy = (y & 1) ? 0.5f : -0.5f; // rel_y = -0.5 even row, +0.5 odd row
  float sx = (x & 1) ? 0.5f : -0.5f;
  const _Float16* arow = (const _Float16*)(s_amp + ((wv >> 1) * 8 + (n >> 1)) * 144);
#pragma unroll
  for (int mb = 0; mb < 4; ++mb) {
#pragma unroll
    for (int t = 0; t < 4; ++t) {
      int c = mb * 8 + hi * 4 + t; // freq pair (2c,2c+1) = rows (2t,2t+1)+8*hi of block mb
      float f = sy * acc[mb][2 * t] + sx * acc[mb][2 * t + 1] + 0.5f * phase_w[c];
      float arg = PI_F * f;
      xt[n * 68 + c]      = (_Float16)((float)arow[c]      * __cosf(arg));
      xt[n * 68 + c + 32] = (_Float16)((float)arow[c + 32] * __sinf(arg));
    }
  }
  asm volatile("s_wait_dscnt 0" ::: "memory");

  // ---- three sin(30 * (W x + b)) layers, 64x64 each ----
  const float* lb[3] = {b0, b1, b2};
#pragma unroll
  for (int l = 0; l < 3; ++l) {
    v16h xf0 = load_frag(xt + n * 68, hi);
    v16h xf1 = load_frag(xt + n * 68 + 32, hi);
    v8f d[4];
#pragma unroll
    for (int mb = 0; mb < 4; ++mb) {
#pragma unroll
      for (int j = 0; j < 8; ++j) d[mb][j] = lb[l][mb * 16 + hi * 8 + j];
      const _Float16* wrow = wL + (size_t)l * 4096 + (mb * 16 + n) * 64;
      d[mb] = wmma_f16(load_frag(wrow, hi), xf0, d[mb]);
      d[mb] = wmma_f16(load_frag(wrow + 32, hi), xf1, d[mb]);
    }
#pragma unroll
    for (int mb = 0; mb < 4; ++mb)
#pragma unroll
      for (int jp = 0; jp < 4; ++jp)
        *(u32*)(xt + n * 68 + mb * 16 + hi * 8 + 2 * jp) =
            pack2h(__sinf(30.0f * d[mb][2 * jp]), __sinf(30.0f * d[mb][2 * jp + 1]));
    asm volatile("s_wait_dscnt 0" ::: "memory");
  }

  // ---- output layer: 12x64 (padded to 16) ----
  {
    v16h xf0 = load_frag(xt + n * 68, hi);
    v16h xf1 = load_frag(xt + n * 68 + 32, hi);
    v8f o;
#pragma unroll
    for (int j = 0; j < 8; ++j) {
      int m = hi * 8 + j;
      o[j] = (m < 12) ? bo[m] : 0.0f;
    }
    o = wmma_f16(load_frag(wO + n * 64, hi), xf0, o);
    o = wmma_f16(load_frag(wO + n * 64 + 32, hi), xf1, o);
#pragma unroll
    for (int j = 0; j < 8; ++j) {
      int m = hi * 8 + j;
      if (m < 12)
        out[(size_t)m * NPIX_HR + (size_t)y * HRW + x] = o[j];
    }
  }
}

extern "C" void kernel_launch(void* const* d_in, const int* in_sizes, int n_in,
                              void* d_out, int out_size, void* d_ws, size_t ws_size,
                              hipStream_t stream) {
  const float* lr_feat = (const float*)d_in[0];
  const float* gb_feat = (const float*)d_in[1];
  const float* amp_w   = (const float*)d_in[2];
  const float* amp_b   = (const float*)d_in[3];
  const float* freq_w  = (const float*)d_in[4];
  const float* freq_b  = (const float*)d_in[5];
  const float* phase_w = (const float*)d_in[6];
  const float* w0 = (const float*)d_in[7];
  const float* b0 = (const float*)d_in[8];
  const float* w1 = (const float*)d_in[9];
  const float* b1 = (const float*)d_in[10];
  const float* w2 = (const float*)d_in[11];
  const float* b2 = (const float*)d_in[12];
  const float* wo = (const float*)d_in[13];
  const float* bo = (const float*)d_in[14];
  float* out = (float*)d_out;

  // workspace layout (~103.3 MB total)
  unsigned char* ws = (unsigned char*)d_ws;
  size_t off = 0;
  _Float16* gbT  = (_Float16*)(ws + off); off += (size_t)NPIX_HR * 32 * 2; // 59.0 MB
  _Float16* lrT  = (_Float16*)(ws + off); off += (size_t)NPIX_LR * 32 * 2; // 14.7 MB
  _Float16* ampT = (_Float16*)(ws + off); off += (size_t)NPIX_LR * 64 * 2; // 29.5 MB
  _Float16* wF   = (_Float16*)(ws + off); off += 9 * 64 * 32 * 2;
  _Float16* wA   = (_Float16*)(ws + off); off += 9 * 64 * 32 * 2;
  _Float16* wLp  = (_Float16*)(ws + off); off += 3 * 64 * 64 * 2;
  _Float16* wOp  = (_Float16*)(ws + off); off += 16 * 64 * 2;
  (void)in_sizes; (void)n_in; (void)out_size; (void)ws_size;

  prep_weights<<<1, 256, 0, stream>>>(amp_w, freq_w, w0, w1, w2, wo, wA, wF, wLp, wOp);
  nchw_to_hwc_f16<<<NPIX_HR / 256, 256, 0, stream>>>(gb_feat, gbT, NPIX_HR);
  nchw_to_hwc_f16<<<NPIX_LR / 256, 256, 0, stream>>>(lr_feat, lrT, NPIX_LR);
  amp_conv<<<45 * 40, 256, 0, stream>>>(amp_b, lrT, wA, ampT);
  inr_pipeline<<<90 * 80, 256, 0, stream>>>(freq_b, phase_w, b0, b1, b2, bo,
                                            gbT, ampT, wF, wLp, wOp, out);
}
